// ModelEncodeDecodeSimple_34368328303268
// MI455X (gfx1250) — matrix operand order repaired and checked by reading, not compiled, not run on hardware
//
#include <hip/hip_runtime.h>

typedef __attribute__((ext_vector_type(16))) _Float16 v16h;
typedef __attribute__((ext_vector_type(8)))  float    v8f;
typedef __attribute__((ext_vector_type(4)))  unsigned int u32x4;
typedef __attribute__((ext_vector_type(8)))  int      i32x8;
typedef __attribute__((ext_vector_type(4)))  int      i32x4;

#define DEVFN __device__ __forceinline__

// ---------------- problem constants ----------------
constexpr int BATCH = 16;
constexpr int NBG   = 60;   // spatial grid per angle
constexpr int LW    = 30;   // number of degrees l
constexpr int MW    = 59;   // centered frequency width (2L-1)
constexpr int PPK   = 6;    // kernel grid points
constexpr int MWP   = 64;   // MW padded to 64
constexpr int KMAX  = 608;  // largest K any GEMM uses
constexpr double PI = 3.14159265358979323846;

// ---------------- workspace layout (bytes) ----------------
constexpr size_t OFF_WQ  = 0;          // f32[64]               quadrature weights
constexpr size_t OFF_WGH = 256;        // f32[2][64]            gamma-integration spectral vec
constexpr size_t OFF_FFT = 768;        // f16[128][64]          forward alpha DFT (re|im rows)
constexpr size_t OFF_B5T = 17152;      // f16[64][128]          inverse alpha DFT (real part)
constexpr size_t OFF_DK  = 33536;      // f32[30][64]           d^l_{n,0}(pi/160)
constexpr size_t OFF_B2T = 41216;      // f16[59][32][64]       wq[j]*d^l_{m,0}(beta_j)
constexpr size_t OFF_B3T = 282880;     // f16[30][48][64]       folded kernel spectrum q
constexpr size_t OFF_DPK = 467200;     // f16[30][60][64][64]   d^l_{m,n}(beta_j)
constexpr size_t OFF_A1  = 15212800;   // f16[9600][64]
constexpr size_t OFF_C1  = 16441600;   // f32[9600][128]        xfc (re|im)
constexpr size_t OFF_A2  = 21356800;   // f16[59][320][64]
constexpr size_t OFF_C2  = 23773440;   // f32[59][320][32]      X spectrum
constexpr size_t OFF_C3  = 26190080;   // f32[1800][64][48]     S raw
constexpr size_t OFF_B4T = 48308480;   // f16 max 9062400 halves
constexpr size_t OFF_A4  = 66433280;   // f16[59][16][608]
constexpr size_t OFF_C4  = 67581184;   // f32[59][16][1200]     y1 spectrum
constexpr size_t OFF_A5  = 72112384;   // f16[9600][128]
constexpr size_t OFF_C5  = 74569984;   // f32[9600][64]         spatial grid (padded cols)

// ---------------- Wigner-d (explicit sum, log-space) ----------------
DEVFN double wigner_d(int l, int mp, int m, double beta) {
  double ch = cos(0.5 * beta), sh = sin(0.5 * beta);
  double lc = log(ch), ls = log(sh);
  double pref = 0.5 * (lgamma((double)(l + mp + 1)) + lgamma((double)(l - mp + 1)) +
                       lgamma((double)(l + m + 1))  + lgamma((double)(l - m + 1)));
  int kmin = (m - mp > 0) ? (m - mp) : 0;
  int kmax = (l + m < l - mp) ? (l + m) : (l - mp);
  double acc = 0.0;
  for (int k = kmin; k <= kmax; ++k) {
    double lt = pref
        - lgamma((double)(l + m - k + 1)) - lgamma((double)(k + 1))
        - lgamma((double)(l - k - mp + 1)) - lgamma((double)(mp - m + k + 1))
        + (double)(2 * l + m - mp - 2 * k) * lc
        + (double)(mp - m + 2 * k) * ls;
    double t = exp(lt);
    acc += ((mp - m + k) & 1) ? -t : t;
  }
  return acc;
}

// ---------------- constant builders ----------------
__global__ void k_const_misc(float* wq, float* wgh, _Float16* fftc, _Float16* b5t) {
  int tid = threadIdx.x;
  for (int j = tid; j < 64; j += 256) {
    float v = 0.f;
    if (j < NBG) {
      double s = 0.0;
      for (int k = 0; k < LW; ++k)
        s += sin(PI * (2 * j + 1) * (2 * k + 1) / (4.0 * LW)) / (2 * k + 1);
      v = (float)((2.0 / LW) * sin(PI * (2 * j + 1) / (4.0 * LW)) * s);
    }
    wq[j] = v;
  }
  __syncthreads();
  for (int m = tid; m < 64; m += 256) {
    double re = 0, im = 0;
    if (m < MW) {
      int cm = m - (LW - 1);
      for (int g = 0; g < NBG; ++g) {
        double th = 2.0 * PI * cm * g / NBG;
        re += wq[g] * cos(th);
        im += wq[g] * sin(th);
      }
    }
    wgh[m] = (float)re;
    wgh[64 + m] = (float)im;
  }
  for (int i = tid; i < 128 * 64; i += 256) {
    int n = i >> 6, a = i & 63;
    float v = 0.f;
    if (a < NBG && n < 2 * MW) {
      int m = (n < MW) ? n : n - MW;
      double th = 2.0 * PI * (m - (LW - 1)) * a / NBG;
      double sc = 2.0 * PI / NBG;
      v = (float)((n < MW) ? cos(th) * sc : -sin(th) * sc);
    }
    fftc[i] = (_Float16)v;
  }
  for (int i = tid; i < 64 * 128; i += 256) {
    int a = i >> 7, k = i & 127;
    float v = 0.f;
    if (a < NBG && k < 2 * MW) {
      int m = k >> 1;
      double th = 2.0 * PI * (m - (LW - 1)) * a / NBG;
      v = (float)((k & 1) ? -sin(th) : cos(th));
    }
    b5t[i] = (_Float16)v;
  }
}

__global__ void k_wigner_dpack(_Float16* dpk) {
  long idx = (long)blockIdx.x * 256 + threadIdx.x;
  const long total = (long)LW * NBG * MWP * MWP;
  if (idx >= total) return;
  int n = idx & 63;
  int m = (int)((idx >> 6) & 63);
  long t = idx >> 12;
  int j = (int)(t % NBG);
  int l = (int)(t / NBG);
  float v = 0.f;
  if (m < MW && n < MW) {
    int cm = m - (LW - 1), cn = n - (LW - 1);
    if (cm >= -l && cm <= l && cn >= -l && cn <= l) {
      double beta = PI * (2 * j + 1) / (4.0 * LW);
      v = (float)wigner_d(l, cm, cn, beta);
    }
  }
  dpk[idx] = (_Float16)v;
}

__global__ void k_wigner_b2t(_Float16* b2t, const float* wq) {
  int idx = blockIdx.x * 256 + threadIdx.x;
  if (idx >= MW * 32 * 64) return;
  int j = idx & 63;
  int l = (idx >> 6) & 31;
  int m = idx >> 11;
  float v = 0.f;
  if (l < LW && j < NBG) {
    int cm = m - (LW - 1);
    if (cm >= -l && cm <= l) {
      double beta = PI * (2 * j + 1) / (4.0 * LW);
      v = wq[j] * (float)wigner_d(l, cm, 0, beta);
    }
  }
  b2t[idx] = (_Float16)v;
}

__global__ void k_wigner_dk(float* dk) {
  int idx = blockIdx.x * 256 + threadIdx.x;
  if (idx >= LW * 64) return;
  int n = idx & 63, l = idx >> 6;
  float v = 0.f;
  if (n < MW) {
    int cn = n - (LW - 1);
    if (cn >= -l && cn <= l) v = (float)wigner_d(l, cn, 0, PI / 160.0);
  }
  dk[idx] = v;
}

// folded kernel spectrum: q = conj(psi) * wgh(n) * (2l+1), packed as GEMM3 B^T
__global__ void k_qpack(_Float16* b3t, const float* w, const float* dk, const float* wgh,
                        int NI, int NO) {
  int idx = blockIdx.x * 256 + threadIdx.x;
  if (idx >= LW * 48 * 64) return;
  int n = idx & 63;
  int row = (idx >> 6) % 48;
  int l = idx / (48 * 64);
  float v = 0.f;
  if (row < 2 * NI * NO && n < MW) {
    int pair = row >> 1, s = row & 1;
    int i = pair / NO, o = pair % NO;
    int cn = n - (LW - 1);
    double pr = 0.0, qi = 0.0;
    for (int p = 0; p < PPK; ++p) {
      double al = 2.0 * PI * p / PPK;
      double wv = w[(i * NO + o) * PPK + p];
      pr += wv * cos(cn * al);
      qi -= wv * sin(cn * al);
    }
    double d = dk[l * 64 + n];
    double psre = d * pr, psim = d * qi;
    double wr = wgh[n], wi = wgh[64 + n];
    double fac = 2 * l + 1;
    double qre = (psre * wr + psim * wi) * fac;
    double qim = (psre * wi - psim * wr) * fac;
    v = (float)(s ? qim : qre);
  }
  b3t[idx] = (_Float16)v;
}

// ---------------- pack / plumbing kernels ----------------
__global__ void k_pack_x(_Float16* dst, const float* src, int rs, long total) {
  long idx = (long)blockIdx.x * 256 + threadIdx.x;
  if (idx >= total) return;
  int k = idx & 63;
  long r = idx >> 6;
  dst[idx] = (_Float16)((k < NBG) ? src[r * rs + k] : 0.f);
}

__global__ void k_pack_xfc(_Float16* a2, const float* c1, int BF) {
  long idx = (long)blockIdx.x * 256 + threadIdx.x;
  long total = (long)MW * 2 * BF * 64;
  if (idx >= total) return;
  int j = idx & 63;
  long t = idx >> 6;
  int M2 = 2 * BF;
  int row = (int)(t % M2);
  int m = (int)(t / M2);
  int r = row / BF, bf = row % BF;
  float v = 0.f;
  if (j < NBG) v = c1[((long)bf * NBG + j) * 128 + (m + MW * r)];
  a2[idx] = (_Float16)v;
}

__global__ void k_pack_a4(_Float16* a4, const float* c2, int NI, int K4) {
  long idx = (long)blockIdx.x * 256 + threadIdx.x;
  long total = (long)MW * 16 * K4;
  if (idx >= total) return;
  int k = (int)(idx % K4);
  long t = idx / K4;
  int b = (int)(t % 16);
  int m = (int)(t / 16);
  int BF = BATCH * NI, M2 = 2 * BF;
  int il = k >> 1, r = k & 1;
  int i = il / LW, l = il % LW;
  float v = 0.f;
  if (i < NI) v = c2[((long)m * M2 + r * BF + (b * NI + i)) * 32 + l];
  a4[idx] = (_Float16)v;
}

__global__ void k_pack_s(_Float16* b4t, const float* c3, int NI, int NO, int K4) {
  int NOJ = NO * NBG, N4 = 2 * NOJ;
  long idx = (long)blockIdx.x * 256 + threadIdx.x;
  long total = (long)MW * N4 * K4;
  if (idx >= total) return;
  int k = (int)(idx % K4);
  long t = idx / K4;
  int row = (int)(t % N4);
  int m = (int)(t / N4);
  int s = row / NOJ, oj = row % NOJ;
  int o = oj / NBG, j = oj % NBG;
  int il = k >> 1, r = k & 1;
  int i = il / LW, l = il % LW;
  float v = 0.f;
  if (i < NI) {
    int pair = i * NO + o;
    const float* sp = c3 + (((long)(l * NBG + j) * MWP + m) * 48 + 2 * pair);
    float sre = sp[0], sim = sp[1];
    v = (s == 0) ? ((r == 0) ? sre : -sim)
                 : ((r == 0) ? sim : sre);
  }
  b4t[idx] = (_Float16)v;
}

__global__ void k_pack_y1(_Float16* a5, const float* c4, int NOJ) {
  long idx = (long)blockIdx.x * 256 + threadIdx.x;
  long total = (long)BATCH * NOJ * 128;
  if (idx >= total) return;
  int k = idx & 127;
  long row = idx >> 7;
  int b = (int)(row / NOJ), oj = (int)(row % NOJ);
  float v = 0.f;
  if (k < 2 * MW) {
    int m = k >> 1, s = k & 1;
    v = c4[((long)m * 16 + b) * (2 * NOJ) + s * NOJ + oj];
  }
  a5[idx] = (_Float16)v;
}

__global__ void k_extract(float* out, const float* c5) {
  int idx = blockIdx.x * 256 + threadIdx.x;
  if (idx >= BATCH * 2 * NBG * NBG) return;
  int a = idx % NBG;
  int r = idx / NBG;
  out[idx] = c5[(long)r * 64 + a];
}

// ---------------- generic batched WMMA GEMM with TDM A-panel staging ----------------
// A: M x K row-major f16; Bt: N x K row-major f16 (B transposed); C: M x N f32.
// M % 16 == 0, N % 16 == 0, K % 32 == 0, K <= KMAX. Grid: (M/16, ceil(N/64), batch); 128 thr.
// The 16xK A panel (shared by all 4 waves of the block) is DMA'd into LDS once by the
// Tensor Data Mover (D# per CDNA5 ISA ch.8), tracked by TENSORcnt; the K loop then runs
// barrier-free: LDS A fragments + streamed global B fragments + v_wmma.
__global__ void k_wmma_gemm(const _Float16* __restrict__ A,
                            const _Float16* __restrict__ Bt,
                            float* __restrict__ C,
                            int M, int N, int K,
                            long sA, long sB, long sC, int bdivB) {
  __shared__ _Float16 As[16 * KMAX];
  const int z = blockIdx.z;
  A  += (long)z * sA;
  Bt += (long)(z / bdivB) * sB;
  C  += (long)z * sC;
  const int lane = threadIdx.x & 31;
  const int wv   = threadIdx.x >> 5;
  const int tm = blockIdx.x;
  int tn = blockIdx.y * 4 + wv;
  const bool active = (tn * 16) < N;
  if (!active) tn = 0;  // clamp; store is guarded (barrier safety)
  const _Float16* Abase = A + (long)(tm * 16) * K;
  const _Float16* Bbase = Bt + (long)(tn * 16) * K;

#if __has_builtin(__builtin_amdgcn_tensor_load_to_lds) && __has_builtin(__builtin_amdgcn_s_wait_tensorcnt)
  if (wv == 0) {
    // Tensor DMA descriptor (D#), ISA ch.8: 2D tile, 16 rows x K f16, row stride K.
    unsigned ldsa = (unsigned)(unsigned long long)(void*)As;       // low 32 bits = LDS addr
    unsigned long long ga = (unsigned long long)(const void*)Abase;
    u32x4 g0;
    g0[0] = 1u;                                                    // count=1, user mode
    g0[1] = ldsa;                                                  // lds_addr
    g0[2] = (unsigned)ga;                                          // global_addr[31:0]
    g0[3] = (unsigned)((ga >> 32) & 0x01ffffffull) | (2u << 30);   // addr[56:32] | type=2
    i32x8 g1;
    g1[0] = 1 << 16;                                               // data_size = 2 bytes
    g1[1] = (int)(((unsigned)K & 0xffffu) << 16);                  // tensor_dim0[15:0]
    g1[2] = (int)((((unsigned)K >> 16) & 0xffffu) | (16u << 16));  // dim0 hi | tensor_dim1 lo
    g1[3] = (int)(((unsigned)K & 0xffffu) << 16);                  // tile_dim0 = K
    g1[4] = 16;                                                    // tile_dim1 = 16 rows
    g1[5] = K;                                                     // tensor_dim0_stride lo32
    g1[6] = 0;
    g1[7] = 0;
    i32x4 gz = {0, 0, 0, 0};
#if defined(__clang_major__) && (__clang_major__ >= 23)
    i32x8 z8 = {0, 0, 0, 0, 0, 0, 0, 0};
    __builtin_amdgcn_tensor_load_to_lds(g0, g1, gz, gz, z8, 0);
#else
    __builtin_amdgcn_tensor_load_to_lds(g0, g1, gz, gz, 0);
#endif
    __builtin_amdgcn_s_wait_tensorcnt(0);
  }
  __syncthreads();
#else
  // fallback: cooperative 128-thread copy of the 16 x K panel
  for (int h0 = threadIdx.x * 8; h0 < 16 * K; h0 += 128 * 8) {
    int r = h0 / K, c = h0 % K;
    const _Float16* gp = Abase + (long)r * K + c;
#pragma unroll
    for (int t = 0; t < 8; ++t) As[h0 + t] = gp[t];
  }
  __syncthreads();
#endif

  v8f acc = {};
  for (int kb = 0; kb < K; kb += 32) {
    v16h fa;
    {  // A fragment: lane<16 -> row=lane, K {0..7,16..23}; lane>=16 -> K {8..15,24..31}
      int row = lane & 15, k0 = (lane >> 4) * 8;
      const _Float16* p = As + (long)row * K + kb + k0;
#pragma unroll
      for (int t = 0; t < 8; ++t) { fa[t] = p[t]; fa[t + 8] = p[16 + t]; }
    }
    v16h fb;
    {  // B fragment from Bt: lane holds column (lane&15), K {0..7,16..23} / {8..15,24..31}
      int col = lane & 15, k0 = (lane >> 4) * 8;
      const _Float16* p = Bbase + (long)col * K + kb + k0;
#pragma unroll
      for (int t = 0; t < 8; ++t) { fb[t] = p[t]; fb[t + 8] = p[16 + t]; }
    }
    if (kb + 32 < K)
      __builtin_prefetch((const void*)(Bbase + (long)(lane & 15) * K + kb + 32), 0, 1);
    acc = __builtin_amdgcn_wmma_f32_16x16x32_f16(false, fa, false, fb,
                                                 (short)0, acc, false, false);
  }
  if (active) {
    int col = tn * 16 + (lane & 15);
    int rbase = (lane >> 4) * 8;
    float* Cp = C + (long)(tm * 16) * N;
#pragma unroll
    for (int r = 0; r < 8; ++r)
      Cp[(long)(rbase + r) * N + col] = acc[r];
  }
}

// ---------------- host launch ----------------
extern "C" void kernel_launch(void* const* d_in, const int* in_sizes, int n_in,
                              void* d_out, int out_size, void* d_ws, size_t ws_size,
                              hipStream_t stream) {
  (void)in_sizes; (void)n_in; (void)out_size; (void)ws_size;
  const float* x1       = (const float*)d_in[0];
  const float* w_conv   = (const float*)d_in[1];
  const float* w_deconv = (const float*)d_in[2];
  float* out = (float*)d_out;
  char* ws = (char*)d_ws;

  float*    WQ   = (float*)(ws + OFF_WQ);
  float*    WGH  = (float*)(ws + OFF_WGH);
  _Float16* FFTC = (_Float16*)(ws + OFF_FFT);
  _Float16* B5T  = (_Float16*)(ws + OFF_B5T);
  float*    DK   = (float*)(ws + OFF_DK);
  _Float16* B2T  = (_Float16*)(ws + OFF_B2T);
  _Float16* B3T  = (_Float16*)(ws + OFF_B3T);
  _Float16* DPK  = (_Float16*)(ws + OFF_DPK);
  _Float16* A1   = (_Float16*)(ws + OFF_A1);
  float*    C1   = (float*)(ws + OFF_C1);
  _Float16* A2   = (_Float16*)(ws + OFF_A2);
  float*    C2   = (float*)(ws + OFF_C2);
  float*    C3   = (float*)(ws + OFF_C3);
  _Float16* B4T  = (_Float16*)(ws + OFF_B4T);
  _Float16* A4   = (_Float16*)(ws + OFF_A4);
  float*    C4   = (float*)(ws + OFF_C4);
  _Float16* A5   = (_Float16*)(ws + OFF_A5);
  float*    C5   = (float*)(ws + OFF_C5);

  auto nb = [](long n) { return (int)((n + 255) / 256); };

  // constants (data-independent, recomputed deterministically every call)
  k_const_misc<<<1, 256, 0, stream>>>(WQ, WGH, FFTC, B5T);
  k_wigner_dpack<<<nb((long)LW * NBG * MWP * MWP), 256, 0, stream>>>(DPK);
  k_wigner_b2t<<<nb((long)MW * 32 * 64), 256, 0, stream>>>(B2T, WQ);
  k_wigner_dk<<<nb(LW * 64), 256, 0, stream>>>(DK);

  for (int phase = 0; phase < 2; ++phase) {
    const int NI = phase ? 10 : 2;
    const int NO = phase ? 2 : 10;
    const float* w    = phase ? w_deconv : w_conv;
    const float* xsrc = phase ? C5 : x1;
    const int rs = phase ? 64 : NBG;

    // Stage 1: forward alpha DFT (centered)  [R1 x 64] x [128 x 64]^T
    const int R1 = BATCH * NI * NBG;
    k_pack_x<<<nb((long)R1 * 64), 256, 0, stream>>>(A1, xsrc, rs, (long)R1 * 64);
    k_wmma_gemm<<<dim3(R1 / 16, 2, 1), 128, 0, stream>>>(
        A1, FFTC, C1, R1, 128, 64, 0, 0, 0, 1);

    // Stage 2: S2 analysis over beta (batched over m)
    const int BF = BATCH * NI, M2 = 2 * BF;
    k_pack_xfc<<<nb((long)MW * M2 * 64), 256, 0, stream>>>(A2, C1, BF);
    k_wmma_gemm<<<dim3(M2 / 16, 1, MW), 128, 0, stream>>>(
        A2, B2T, C2, M2, 32, 64, (long)M2 * 64, 32 * 64, (long)M2 * 32, 1);

    // Stage 3: S[i,o,l,m,j] = sum_n d^l_{mn}(beta_j) * q  (batched over (l,j))
    k_qpack<<<nb((long)LW * 48 * 64), 256, 0, stream>>>(B3T, w, DK, WGH, NI, NO);
    k_wmma_gemm<<<dim3(4, 1, LW * NBG), 128, 0, stream>>>(
        DPK, B3T, C3, 64, 48, 64, 4096, 48 * 64, 64 * 48, 60);

    // Stage 4: y1[b,o,j,m] = sum_{i,l} X * S  (complex via real GEMM, batched over m)
    const int K4 = ((2 * NI * LW + 31) / 32) * 32;
    const int NOJ = NO * NBG, N4 = 2 * NOJ;
    k_pack_a4<<<nb((long)MW * 16 * K4), 256, 0, stream>>>(A4, C2, NI, K4);
    k_pack_s<<<nb((long)MW * N4 * K4), 256, 0, stream>>>(B4T, C3, NI, NO, K4);
    k_wmma_gemm<<<dim3(1, (N4 + 63) / 64, MW), 128, 0, stream>>>(
        A4, B4T, C4, 16, N4, K4, (long)16 * K4, (long)N4 * K4, (long)16 * N4, 1);

    // Stage 5: inverse alpha DFT, real part (gamma already integrated analytically)
    const int R5 = BATCH * NOJ;
    k_pack_y1<<<nb((long)R5 * 128), 256, 0, stream>>>(A5, C4, NOJ);
    k_wmma_gemm<<<dim3(R5 / 16, 1, 1), 128, 0, stream>>>(
        A5, B5T, C5, R5, 64, 128, 0, 0, 0, 1);
  }

  k_extract<<<nb((long)BATCH * 2 * NBG * NBG), 256, 0, stream>>>(out, C5);
}
